// MultiHeadAttention_81647328297132
// MI455X (gfx1250) — matrix operand order repaired and checked by reading, not compiled, not run on hardware
//
#include <hip/hip_runtime.h>
#include <hip/hip_bf16.h>

// ---------------------------------------------------------------------------
// GQA attention for MI455X (gfx1250), wave32 + v_wmma_f32_16x16x32_bf16.
// cvt X/W -> bf16, software-pipelined WMMA GEMM (Q,K,V proj), flash-attention
// with pipelined K/V fragment loads, WMMA GEMM output projection (fp32 out).
// ---------------------------------------------------------------------------

typedef __attribute__((ext_vector_type(16))) __bf16        v16bf;
typedef __attribute__((ext_vector_type(8)))  float         v8f;
typedef __attribute__((ext_vector_type(8)))  unsigned int  v8u;

#define D_MODEL 576
#define N_HEADS 9
#define HEAD_DIM 64
#define N_KV 3
#define KV_DIM (N_KV * HEAD_DIM)   // 192
#define SEQ 2048
#define BATCH 4
#define MTOT (BATCH * SEQ)         // 8192

__device__ __forceinline__ unsigned short f2bf(float f) {
  union { float f; unsigned int u; } c; c.f = f;
  unsigned int u = c.u;
  u += 0x7FFFu + ((u >> 16) & 1u);           // round-to-nearest-even
  return (unsigned short)(u >> 16);
}

// A fragment (16x32 bf16): lane half0 K{0..15}, half1 K{16..31}
__device__ __forceinline__ v8u load_frag_a(const unsigned short* __restrict__ row,
                                           int k0, int hb) {
  v8u u;
#pragma unroll
  for (int j = 0; j < 8; ++j) {
    int kk = (hb << 4) + (j << 1);
    u[j] = *(const unsigned int*)(row + k0 + kk);
  }
  return u;
}

// B fragment (32x16 bf16): ptr already includes column row-base + hb*16 koff
__device__ __forceinline__ v8u load_frag_b(const unsigned short* __restrict__ p) {
  v8u u;
#pragma unroll
  for (int j = 0; j < 8; ++j) u[j] = *(const unsigned int*)(p + (j << 1));
  return u;
}

__device__ __forceinline__ v8f wmma_bf16(v8u a, v8u b, v8f c) {
  return __builtin_amdgcn_wmma_f32_16x16x32_bf16(
      false, __builtin_bit_cast(v16bf, a), false, __builtin_bit_cast(v16bf, b),
      (short)0, c, false, false);
}

// ---------------------------------------------------------------------------
// elementwise fp32 -> bf16
__global__ void cvt_bf16_kernel(const float* __restrict__ x,
                                unsigned short* __restrict__ y, int n) {
  int i = blockIdx.x * blockDim.x + threadIdx.x;
  if (i < n) y[i] = f2bf(x[i]);
}

// W (K x N, row-major fp32) -> Wt (N x K, row-major bf16)
__global__ void transpose_bf16_kernel(const float* __restrict__ w,
                                      unsigned short* __restrict__ wt,
                                      int K, int N) {
  int i = blockIdx.x * blockDim.x + threadIdx.x;
  if (i >= K * N) return;
  int n = i / K, k = i - n * K;
  wt[i] = f2bf(w[(size_t)k * N + n]);
}

// ---------------------------------------------------------------------------
// WMMA GEMM: out(MxN) = A(MxK,bf16) * Bt(NxK,bf16)^T + bias
// One wave computes a 32x64 tile, software-pipelined (depth 1).
// mode: 0 = bf16 row-major, 1 = bf16 V-transpose [b][n][s], 2 = fp32 row-major.
__global__ __launch_bounds__(256)
void gemm_wmma_bf16(const unsigned short* __restrict__ A,
                    const unsigned short* __restrict__ Bt,
                    const float* __restrict__ bias,
                    void* __restrict__ out,
                    int M, int N, int K, int mode) {
  const int wid  = (blockIdx.x * blockDim.x + threadIdx.x) >> 5;
  const int lane = threadIdx.x & 31;
  const int nChunks = N >> 6;
  if (wid >= (M >> 5) * nChunks) return;
  const int nt = wid % nChunks;
  const int mt = wid / nChunks;
  const int m0 = mt << 5, n0 = nt << 6;
  const int ln = lane & 15, hb = lane >> 4;

  v8f c[2][4];
#pragma unroll
  for (int r = 0; r < 2; ++r)
#pragma unroll
    for (int t = 0; t < 4; ++t)
#pragma unroll
      for (int i = 0; i < 8; ++i) c[r][t][i] = 0.0f;

  const unsigned short* arow0 = A + (size_t)(m0 + ln) * K;
  const unsigned short* arow1 = A + (size_t)(m0 + 16 + ln) * K;
  const unsigned short* bcol[4];
#pragma unroll
  for (int t = 0; t < 4; ++t)
    bcol[t] = Bt + (size_t)(n0 + t * 16 + ln) * K + (hb << 4);

  // prologue
  v8u a0c = load_frag_a(arow0, 0, hb);
  v8u a1c = load_frag_a(arow1, 0, hb);
  v8u bc[4];
#pragma unroll
  for (int t = 0; t < 4; ++t) bc[t] = load_frag_b(bcol[t]);

  for (int k0 = 32; k0 < K; k0 += 32) {
    // issue next k-step's loads before this step's WMMAs
    v8u a0n = load_frag_a(arow0, k0, hb);
    v8u a1n = load_frag_a(arow1, k0, hb);
    v8u bn[4];
#pragma unroll
    for (int t = 0; t < 4; ++t) bn[t] = load_frag_b(bcol[t] + k0);
#pragma unroll
    for (int t = 0; t < 4; ++t) {
      c[0][t] = wmma_bf16(a0c, bc[t], c[0][t]);
      c[1][t] = wmma_bf16(a1c, bc[t], c[1][t]);
    }
    a0c = a0n; a1c = a1n;
#pragma unroll
    for (int t = 0; t < 4; ++t) bc[t] = bn[t];
  }
  // epilogue
#pragma unroll
  for (int t = 0; t < 4; ++t) {
    c[0][t] = wmma_bf16(a0c, bc[t], c[0][t]);
    c[1][t] = wmma_bf16(a1c, bc[t], c[1][t]);
  }

#pragma unroll
  for (int r = 0; r < 2; ++r) {
#pragma unroll
    for (int t = 0; t < 4; ++t) {
      const int col = n0 + t * 16 + ln;
      const float bv = bias[col];
#pragma unroll
      for (int i = 0; i < 8; ++i) {
        const int row = m0 + r * 16 + i + (hb << 3);
        const float v = c[r][t][i] + bv;
        if (mode == 0) {
          ((unsigned short*)out)[(size_t)row * N + col] = f2bf(v);
        } else if (mode == 1) {
          const int b = row >> 11, s = row & (SEQ - 1);
          ((unsigned short*)out)[((size_t)(b * KV_DIM + col) << 11) + s] =
              f2bf(v);
        } else {
          ((float*)out)[(size_t)row * N + col] = v;
        }
      }
    }
  }
}

// ---------------------------------------------------------------------------
// Flash attention: one wave per (b, head, 16-row q tile); 32-key blocks.
// K fragments pipelined one block ahead; V fragment loads issued before the
// softmax VALU work so their latency hides behind it.
// Qbf: [b*S + s][9*64] bf16    Kbf: [b*S + s][3*64] bf16
// Vt : [b][kv*64+d][s]  bf16   Obf: [b*S + s][9*64] bf16
__global__ __launch_bounds__(128)
void flash_attn_wmma(const unsigned short* __restrict__ Qbf,
                     const unsigned short* __restrict__ Kbf,
                     const unsigned short* __restrict__ Vt,
                     unsigned short* __restrict__ Obf) {
  __shared__ __align__(16) unsigned short pbuf[4][16 * 32];
  const int wslot = threadIdx.x >> 5;
  const int wid   = (blockIdx.x * blockDim.x + threadIdx.x) >> 5;
  if (wid >= BATCH * N_HEADS * (SEQ / 16)) return;
  const int lane = threadIdx.x & 31;
  const int ln = lane & 15, hb = lane >> 4;
  const int qt = wid & (SEQ / 16 - 1);
  const int h  = (wid >> 7) % N_HEADS;
  const int b  = wid / ((SEQ / 16) * N_HEADS);
  const int q0 = qt << 4;
  const int kvh = h / (N_HEADS / N_KV);

  // Q A-fragments for two d k-steps (d 0..31 and 32..63)
  const unsigned short* qrow =
      Qbf + (size_t)(b * SEQ + q0 + ln) * D_MODEL + h * HEAD_DIM;
  v8u aq[2];
#pragma unroll
  for (int s = 0; s < 2; ++s) aq[s] = load_frag_a(qrow, s * 32, hb);

  v8f o[4];
#pragma unroll
  for (int t = 0; t < 4; ++t)
#pragma unroll
    for (int i = 0; i < 8; ++i) o[t][i] = 0.0f;
  float m[8], l[8];
#pragma unroll
  for (int i = 0; i < 8; ++i) { m[i] = -3.0e38f; l[i] = 0.0f; }

  const int kend = q0 + 15;
  const float sc = 0.125f;  // 1/sqrt(64)
  unsigned short* pw = pbuf[wslot];

  // K fragment base for (key row, d-step): kf[tile][st]
  const unsigned short* kbase =
      Kbf + (size_t)(b * SEQ + ln) * KV_DIM + kvh * HEAD_DIM + (hb << 4);
  // prologue: K fragments for kb = 0
  v8u kf[2][2];
#pragma unroll
  for (int kt = 0; kt < 2; ++kt)
#pragma unroll
    for (int st = 0; st < 2; ++st)
      kf[kt][st] = load_frag_b(kbase + (size_t)(kt * 16) * KV_DIM + st * 32);

  for (int kb = 0; kb <= kend; kb += 32) {
    // ----- scores: two 16x16 key tiles (K frags already in registers)
    v8f s0, s1;
#pragma unroll
    for (int i = 0; i < 8; ++i) { s0[i] = 0.0f; s1[i] = 0.0f; }
#pragma unroll
    for (int st = 0; st < 2; ++st) {
      s0 = wmma_bf16(aq[st], kf[0][st], s0);
      s1 = wmma_bf16(aq[st], kf[1][st], s1);
    }

    // ----- issue V fragment loads for this block (independent of softmax)
    v8u vb[4];
#pragma unroll
    for (int t = 0; t < 4; ++t) {
      const unsigned short* vp =
          Vt + ((size_t)(b * KV_DIM + kvh * HEAD_DIM + t * 16 + ln) << 11) +
          kb + (hb << 4);
      vb[t] = load_frag_b(vp);
    }
    // ----- issue next block's K fragment loads (clamped in-bounds)
    const int kbn = (kb + 32 <= kend) ? kb + 32 : kb;
    v8u kn[2][2];
#pragma unroll
    for (int kt = 0; kt < 2; ++kt)
#pragma unroll
      for (int st = 0; st < 2; ++st)
        kn[kt][st] =
            load_frag_b(kbase + (size_t)(kbn + kt * 16) * KV_DIM + st * 32);
    if (kb + 64 <= kend)  // pull the block after that toward the WGP
      __builtin_prefetch(Kbf + (size_t)(b * SEQ + kb + 64 + ln) * KV_DIM +
                             kvh * HEAD_DIM, 0, 0);

    // ----- scale + causal mask (C layout: q = q0 + slot + 8*hb, key = kb + ln)
    const bool full = (kb + 31 <= q0);
#pragma unroll
    for (int i = 0; i < 8; ++i) {
      const int q = q0 + i + (hb << 3);
      float v0 = s0[i] * sc, v1 = s1[i] * sc;
      if (!full) {
        if (kb + ln > q)      v0 = -3.0e38f;
        if (kb + 16 + ln > q) v1 = -3.0e38f;
      }
      s0[i] = v0; s1[i] = v1;
    }

    // ----- online softmax (row reductions across 16-lane groups)
    float scl[8];
#pragma unroll
    for (int i = 0; i < 8; ++i) {
      float rm = fmaxf(s0[i], s1[i]);
#pragma unroll
      for (int off = 1; off < 16; off <<= 1)
        rm = fmaxf(rm, __shfl_xor(rm, off, 32));
      const float mn = fmaxf(m[i], rm);
      const float p0 = __expf(s0[i] - mn);
      const float p1 = __expf(s1[i] - mn);
      float rs = p0 + p1;
#pragma unroll
      for (int off = 1; off < 16; off <<= 1) rs += __shfl_xor(rs, off, 32);
      const float so = __expf(m[i] - mn);
      l[i] = l[i] * so + rs;
      m[i] = mn;
      scl[i] = so;
      s0[i] = p0; s1[i] = p1;
    }
#pragma unroll
    for (int t = 0; t < 4; ++t)
#pragma unroll
      for (int i = 0; i < 8; ++i) o[t][i] *= scl[i];

    // ----- P: C-layout -> LDS -> A-layout (per-wave buffer, DS is in-order)
    __asm__ volatile("s_wait_dscnt 0" ::: "memory");
#pragma unroll
    for (int i = 0; i < 8; ++i) {
      const int r = i + (hb << 3);
      pw[r * 32 + ln]      = f2bf(s0[i]);
      pw[r * 32 + 16 + ln] = f2bf(s1[i]);
    }
    __asm__ volatile("s_wait_dscnt 0" ::: "memory");
    v8u pu;
#pragma unroll
    for (int j = 0; j < 8; ++j) {
      int kk = (hb << 4) + (j << 1);
      pu[j] = *(const unsigned int*)(pw + ln * 32 + kk);
    }

    // ----- O += P (16x32) x V (32x64)
#pragma unroll
    for (int t = 0; t < 4; ++t) o[t] = wmma_bf16(pu, vb[t], o[t]);

    // rotate pipelined K fragments
#pragma unroll
    for (int kt = 0; kt < 2; ++kt)
#pragma unroll
      for (int st = 0; st < 2; ++st) kf[kt][st] = kn[kt][st];
  }

  // ----- normalize + store bf16
#pragma unroll
  for (int t = 0; t < 4; ++t)
#pragma unroll
    for (int i = 0; i < 8; ++i) {
      const int q = q0 + i + (hb << 3);
      Obf[(size_t)(b * SEQ + q) * D_MODEL + h * HEAD_DIM + t * 16 + ln] =
          f2bf(o[t][i] / l[i]);
    }
}

// ---------------------------------------------------------------------------
extern "C" void kernel_launch(void* const* d_in, const int* in_sizes, int n_in,
                              void* d_out, int out_size, void* d_ws,
                              size_t ws_size, hipStream_t stream) {
  (void)in_sizes; (void)n_in; (void)out_size; (void)ws_size;
  const float* X  = (const float*)d_in[0];
  const float* Wq = (const float*)d_in[1];
  const float* bq = (const float*)d_in[2];
  const float* Wk = (const float*)d_in[3];
  const float* bk = (const float*)d_in[4];
  const float* Wv = (const float*)d_in[5];
  const float* bv = (const float*)d_in[6];
  const float* Wo = (const float*)d_in[7];
  const float* bo = (const float*)d_in[8];

  // workspace carve-up (bf16 elements); total ~36.4 MB
  unsigned short* w = (unsigned short*)d_ws;
  unsigned short* Xbf  = w;                 // 8192*576
  unsigned short* Qbf  = Xbf  + MTOT * D_MODEL;
  unsigned short* Kbf  = Qbf  + MTOT * D_MODEL;   // 8192*192
  unsigned short* Vt   = Kbf  + MTOT * KV_DIM;    // 4*192*2048
  unsigned short* Obf  = Vt   + BATCH * KV_DIM * SEQ;
  unsigned short* Wqt  = Obf  + MTOT * D_MODEL;   // 576*576
  unsigned short* Wkt  = Wqt  + D_MODEL * D_MODEL;
  unsigned short* Wvt  = Wkt  + D_MODEL * KV_DIM;
  unsigned short* Wot  = Wvt  + D_MODEL * KV_DIM;

  const int nX = MTOT * D_MODEL;
  cvt_bf16_kernel<<<(nX + 255) / 256, 256, 0, stream>>>(X, Xbf, nX);
  transpose_bf16_kernel<<<(D_MODEL * D_MODEL + 255) / 256, 256, 0, stream>>>(
      Wq, Wqt, D_MODEL, D_MODEL);
  transpose_bf16_kernel<<<(D_MODEL * KV_DIM + 255) / 256, 256, 0, stream>>>(
      Wk, Wkt, D_MODEL, KV_DIM);
  transpose_bf16_kernel<<<(D_MODEL * KV_DIM + 255) / 256, 256, 0, stream>>>(
      Wv, Wvt, D_MODEL, KV_DIM);
  transpose_bf16_kernel<<<(D_MODEL * D_MODEL + 255) / 256, 256, 0, stream>>>(
      Wo, Wot, D_MODEL, D_MODEL);

  {  // Q projection: 8192x576 = X @ Wq + bq  (bf16 out)
    const int waves = (MTOT / 32) * (D_MODEL / 64);
    gemm_wmma_bf16<<<(waves + 7) / 8, 256, 0, stream>>>(
        Xbf, Wqt, bq, Qbf, MTOT, D_MODEL, D_MODEL, 0);
  }
  {  // K projection: 8192x192 (bf16 out)
    const int waves = (MTOT / 32) * (KV_DIM / 64);
    gemm_wmma_bf16<<<(waves + 7) / 8, 256, 0, stream>>>(
        Xbf, Wkt, bk, Kbf, MTOT, KV_DIM, D_MODEL, 0);
  }
  {  // V projection, stored transposed [b][kv*64+d][s] (bf16 out)
    const int waves = (MTOT / 32) * (KV_DIM / 64);
    gemm_wmma_bf16<<<(waves + 7) / 8, 256, 0, stream>>>(
        Xbf, Wvt, bv, Vt, MTOT, KV_DIM, D_MODEL, 1);
  }

  // flash attention: 4608 waves, 4 waves/block
  flash_attn_wmma<<<(BATCH * N_HEADS * (SEQ / 16)) / 4, 128, 0, stream>>>(
      Qbf, Kbf, Vt, Obf);

  {  // output projection: fp32 to d_out
    const int waves = (MTOT / 32) * (D_MODEL / 64);
    gemm_wmma_bf16<<<(waves + 7) / 8, 256, 0, stream>>>(
        Obf, Wot, bo, d_out, MTOT, D_MODEL, D_MODEL, 2);
  }
}
